// HeteroDirectionPredictor_78769700208715
// MI455X (gfx1250) — matrix operand order repaired and checked by reading, not compiled, not run on hardware
//
#include <hip/hip_runtime.h>
#include <hip/hip_bf16.h>
#include <cstdint>

#define HDIM 128

typedef __attribute__((ext_vector_type(16))) __bf16        v16bf;
typedef __attribute__((ext_vector_type(8)))  float         v8f;
typedef __attribute__((ext_vector_type(8)))  unsigned int  v8u;

__device__ __forceinline__ unsigned short f2bf(float f) {
  union { float f; unsigned int u; } c; c.f = f;
  unsigned int u = c.u;
  unsigned int r = u + 0x7FFFu + ((u >> 16) & 1u);   // round-to-nearest-even
  return (unsigned short)(r >> 16);
}

// ---------------- weight conversion (once, tiny) ----------------
// Wt3 : [3][128][128] bf16, N-major (transposed) so B-fragment K runs are contiguous
// Wc1t: [128][256]    bf16, N-major (transposed)
__global__ void k_convert_W(const float* __restrict__ Ws,
                            const float* __restrict__ Wo,
                            const float* __restrict__ Wst,
                            const float* __restrict__ Wc1,
                            unsigned short* __restrict__ Wt3,
                            unsigned short* __restrict__ Wc1t) {
  int i = blockIdx.x * blockDim.x + threadIdx.x;
  const int total3 = 3 * HDIM * HDIM;
  if (i < total3) {
    int mat = i / (HDIM * HDIM);
    int r   = i - mat * HDIM * HDIM;
    int n   = r >> 7;          // output row (= GEMM column)
    int k   = r & (HDIM - 1);
    const float* W = (mat == 0) ? Ws : (mat == 1) ? Wo : Wst;
    Wt3[i] = f2bf(W[k * HDIM + n]);
  } else {
    int r = i - total3;
    if (r < 2 * HDIM * HDIM) {
      int n = r / (2 * HDIM);
      int k = r - n * (2 * HDIM);
      Wc1t[n * (2 * HDIM) + k] = f2bf(Wc1[k * HDIM + n]);
    }
  }
}

// ---------------- h = emb[node_ids] -> bf16 ----------------
__global__ void k_convert_h(const int* __restrict__ node_ids,
                            const float* __restrict__ emb,
                            unsigned short* __restrict__ hbf, int n) {
  int i = blockIdx.x * blockDim.x + threadIdx.x;
  if (i >= n * HDIM) return;
  int row = i >> 7, col = i & (HDIM - 1);
  hbf[i] = f2bf(emb[(size_t)node_ids[row] * HDIM + col]);
}

// ---------------- fused 3-way GEMM: h(N x128) @ W(128x128) ----------------
// mat 0 -> hacc (f32, seed of the aggregation); mat 1/2 -> bf16 side buffers
// (read-only gather sources for the edge scatter: halves the dominant traffic).
__global__ void k_gemm_h3(const unsigned short* __restrict__ hbf,
                          const unsigned short* __restrict__ Wt3,
                          float* __restrict__ hacc,
                          unsigned short* __restrict__ homni16,
                          unsigned short* __restrict__ hstr16,
                          int n) {
  int mat  = blockIdx.y;
  int m0   = blockIdx.x * 16;
  int wave = threadIdx.x >> 5;
  int lane = threadIdx.x & 31;
  int half = lane >> 4;
  int l15  = lane & 15;
  int n0   = wave * 16;
  const unsigned short* Wt = Wt3 + (size_t)mat * HDIM * HDIM;

  int mrow = m0 + l15; if (mrow >= n) mrow = n - 1;        // clamp, keep EXEC full
  const unsigned short* arow = hbf + (size_t)mrow * HDIM;
  const unsigned short* brow = Wt + (size_t)(n0 + l15) * HDIM;

  v8f acc = {};
#pragma unroll
  for (int kc = 0; kc < 4; ++kc) {
    int kbase = kc * 32;
    uint4 a_lo = *(const uint4*)(arow + kbase + 8 * half);
    uint4 a_hi = *(const uint4*)(arow + kbase + 8 * half + 16);
    v8u au = {a_lo.x, a_lo.y, a_lo.z, a_lo.w, a_hi.x, a_hi.y, a_hi.z, a_hi.w};
    uint4 b_lo = *(const uint4*)(brow + kbase + 8 * half);
    uint4 b_hi = *(const uint4*)(brow + kbase + 8 * half + 16);
    v8u bu = {b_lo.x, b_lo.y, b_lo.z, b_lo.w, b_hi.x, b_hi.y, b_hi.z, b_hi.w};
    acc = __builtin_amdgcn_wmma_f32_16x16x32_bf16(
        false, __builtin_bit_cast(v16bf, au),
        false, __builtin_bit_cast(v16bf, bu),
        (short)0, acc, false, false);
  }

  const bool full = (m0 + 16 <= n);                        // block-uniform fast path
  size_t base = (size_t)(m0 + 8 * half) * HDIM + n0 + l15;
  if (mat == 0) {
    if (full) {
#pragma unroll
      for (int r = 0; r < 8; ++r) hacc[base + (size_t)r * HDIM] = acc[r];
    } else {
#pragma unroll
      for (int r = 0; r < 8; ++r)
        if (m0 + r + 8 * half < n) hacc[base + (size_t)r * HDIM] = acc[r];
    }
  } else {
    unsigned short* o16 = (mat == 1) ? homni16 : hstr16;
    if (full) {
#pragma unroll
      for (int r = 0; r < 8; ++r) o16[base + (size_t)r * HDIM] = f2bf(acc[r]);
    } else {
#pragma unroll
      for (int r = 0; r < 8; ++r)
        if (m0 + r + 8 * half < n) o16[base + (size_t)r * HDIM] = f2bf(acc[r]);
    }
  }
}

// ---------------- edge scatter: hacc[dst] += bf16row[src], one wave per edge ----------------
__global__ void k_edge_scatter(const int* __restrict__ edges,   // [E][2] (dst, src)
                               const unsigned short* __restrict__ hsrc16,
                               float* __restrict__ hacc, int E) {
  int e = blockIdx.x * 8 + (threadIdx.x >> 5);
  if (e >= E) return;
  int lane = threadIdx.x & 31;
  int dst = edges[2 * e + 0];
  int src = edges[2 * e + 1];
  const uint2 u = *(const uint2*)(hsrc16 + (size_t)src * HDIM + lane * 4);
  float f0 = __uint_as_float(u.x << 16);                   // bf16 -> f32 (exact)
  float f1 = __uint_as_float(u.x & 0xFFFF0000u);
  float f2 = __uint_as_float(u.y << 16);
  float f3 = __uint_as_float(u.y & 0xFFFF0000u);
  float* p = hacc + (size_t)dst * HDIM + lane * 4;
  asm volatile("global_atomic_add_f32 %0, %1, off" :: "v"(p + 0), "v"(f0) : "memory");
  asm volatile("global_atomic_add_f32 %0, %1, off" :: "v"(p + 1), "v"(f1) : "memory");
  asm volatile("global_atomic_add_f32 %0, %1, off" :: "v"(p + 2), "v"(f2) : "memory");
  asm volatile("global_atomic_add_f32 %0, %1, off" :: "v"(p + 3), "v"(f3) : "memory");
}

// ---------------- relu + bf16 convert ----------------
__global__ void k_relu_cvt(const float* __restrict__ hacc,
                           unsigned short* __restrict__ hnext, int n) {
  int i = blockIdx.x * blockDim.x + threadIdx.x;
  if (i >= n * HDIM) return;
  float f = hacc[i];
  hnext[i] = f2bf(f > 0.f ? f : 0.f);
}

// ---------------- query head: [h_u|h_v](16x256) @ Wc1t -> relu -> @Wc2 -> softmax ----------------
__global__ void k_query(const int* __restrict__ qe,             // [Q][2]
                        const unsigned short* __restrict__ hnext,
                        const unsigned short* __restrict__ Wc1t, // [128][256]
                        const float* __restrict__ bc1,
                        const float* __restrict__ Wc2,           // [128][4]
                        const float* __restrict__ bc2,
                        float* __restrict__ out, int Q) {
  __shared__ float zsh[16][HDIM + 1];                      // +1 pad: kill bank conflicts in head
  __shared__ float lsh[16][4];
  int q0   = blockIdx.x * 16;
  int wave = threadIdx.x >> 5;
  int lane = threadIdx.x & 31;
  int half = lane >> 4;
  int l15  = lane & 15;
  int n0   = wave * 16;

  int q = q0 + l15; if (q >= Q) q = Q - 1;                 // clamp, keep EXEC full
  int u = qe[2 * q + 0];
  int v = qe[2 * q + 1];
  const unsigned short* urow = hnext + (size_t)u * HDIM;
  const unsigned short* vrow = hnext + (size_t)v * HDIM;
  const unsigned short* brow = Wc1t + (size_t)(n0 + l15) * (2 * HDIM);

  v8f acc = {};
#pragma unroll
  for (int kc = 0; kc < 8; ++kc) {
    int kbase = kc * 32;                                   // concat dim: 0..255
    const unsigned short* arow = (kbase < HDIM) ? urow : vrow;
    int koff = kbase & (HDIM - 1);
    uint4 a_lo = *(const uint4*)(arow + koff + 8 * half);
    uint4 a_hi = *(const uint4*)(arow + koff + 8 * half + 16);
    v8u au = {a_lo.x, a_lo.y, a_lo.z, a_lo.w, a_hi.x, a_hi.y, a_hi.z, a_hi.w};
    uint4 b_lo = *(const uint4*)(brow + kbase + 8 * half);
    uint4 b_hi = *(const uint4*)(brow + kbase + 8 * half + 16);
    v8u bu = {b_lo.x, b_lo.y, b_lo.z, b_lo.w, b_hi.x, b_hi.y, b_hi.z, b_hi.w};
    acc = __builtin_amdgcn_wmma_f32_16x16x32_bf16(
        false, __builtin_bit_cast(v16bf, au),
        false, __builtin_bit_cast(v16bf, bu),
        (short)0, acc, false, false);
  }
  float bias = bc1[n0 + l15];
#pragma unroll
  for (int r = 0; r < 8; ++r) {
    float z = acc[r] + bias;
    zsh[r + 8 * half][n0 + l15] = z > 0.f ? z : 0.f;
  }
  __syncthreads();

  int t = threadIdx.x;
  if (t < 64) {                                            // 16 queries x 4 classes
    int m = t >> 2, c = t & 3;
    float dot = bc2[c];
#pragma unroll 8
    for (int k = 0; k < HDIM; ++k) dot += zsh[m][k] * Wc2[k * 4 + c];
    lsh[m][c] = dot;
  }
  __syncthreads();
  if (t < 64) {
    int m = t >> 2, c = t & 3;
    float l0 = lsh[m][0], l1 = lsh[m][1], l2 = lsh[m][2], l3 = lsh[m][3];
    float mx = fmaxf(fmaxf(l0, l1), fmaxf(l2, l3));
    float e0 = __expf(l0 - mx), e1 = __expf(l1 - mx), e2 = __expf(l2 - mx), e3 = __expf(l3 - mx);
    float s = e0 + e1 + e2 + e3;
    float ec = (c == 0) ? e0 : (c == 1) ? e1 : (c == 2) ? e2 : e3;
    if (q0 + m < Q) out[(size_t)(q0 + m) * 4 + c] = ec / s;
  }
}

extern "C" void kernel_launch(void* const* d_in, const int* in_sizes, int n_in,
                              void* d_out, int out_size, void* d_ws, size_t ws_size,
                              hipStream_t stream) {
  const int*   node_ids = (const int*)d_in[0];
  const int*   omni     = (const int*)d_in[1];
  const int*   sedg     = (const int*)d_in[2];
  const int*   qedg     = (const int*)d_in[3];
  const float* emb      = (const float*)d_in[4];
  const float* Wself    = (const float*)d_in[5];
  const float* Womni    = (const float*)d_in[6];
  const float* Wstr     = (const float*)d_in[7];
  const float* Wc1      = (const float*)d_in[8];
  const float* bc1      = (const float*)d_in[9];
  const float* Wc2      = (const float*)d_in[10];
  const float* bc2      = (const float*)d_in[11];
  float* out = (float*)d_out;

  const int n  = in_sizes[0];
  const int E1 = in_sizes[1] / 2;
  const int E2 = in_sizes[2] / 2;
  const int Q  = in_sizes[3] / 2;

  char* ws = (char*)d_ws;
  size_t off = 0;
  auto alloc = [&](size_t bytes) -> char* {
    char* p = ws + off;
    off += (bytes + 255) & ~(size_t)255;
    return p;
  };
  unsigned short* Wt3     = (unsigned short*)alloc((size_t)3 * HDIM * HDIM * 2);
  unsigned short* Wc1t    = (unsigned short*)alloc((size_t)HDIM * 2 * HDIM * 2);
  unsigned short* hbf     = (unsigned short*)alloc((size_t)n * HDIM * 2);
  unsigned short* hnx     = (unsigned short*)alloc((size_t)n * HDIM * 2);
  unsigned short* homni16 = (unsigned short*)alloc((size_t)n * HDIM * 2);
  unsigned short* hstr16  = (unsigned short*)alloc((size_t)n * HDIM * 2);
  float* hacc = (float*)alloc((size_t)n * HDIM * 4);
  (void)ws_size; (void)n_in; (void)out_size;

  {
    int tot = 3 * HDIM * HDIM + 2 * HDIM * HDIM;
    k_convert_W<<<(tot + 255) / 256, 256, 0, stream>>>(Wself, Womni, Wstr, Wc1, Wt3, Wc1t);
  }
  {
    int tot = n * HDIM;
    k_convert_h<<<(tot + 255) / 256, 256, 0, stream>>>(node_ids, emb, hbf, n);
  }
  {
    dim3 grid((n + 15) / 16, 3);
    k_gemm_h3<<<grid, 256, 0, stream>>>(hbf, Wt3, hacc, homni16, hstr16, n);
  }
  k_edge_scatter<<<(E1 + 7) / 8, 256, 0, stream>>>(omni, homni16, hacc, E1);
  k_edge_scatter<<<(E2 + 7) / 8, 256, 0, stream>>>(sedg, hstr16, hacc, E2);
  {
    int tot = n * HDIM;
    k_relu_cvt<<<(tot + 255) / 256, 256, 0, stream>>>(hacc, hnx, n);
  }
  k_query<<<(Q + 15) / 16, 256, 0, stream>>>(qedg, hnx, Wc1t, bc1, Wc2, bc2, out, Q);
}
